// StochasticAttention_48713519071915
// MI455X (gfx1250) — matrix operand order repaired and checked by reading, not compiled, not run on hardware
//
#include <hip/hip_runtime.h>
#include <stdint.h>

typedef __attribute__((ext_vector_type(16))) __bf16        v16bf;
typedef __attribute__((ext_vector_type(8)))  float         v8f;
typedef __attribute__((ext_vector_type(4)))  float         f32x4;
typedef __attribute__((ext_vector_type(2)))  float         f32x2;

#define B_   32
#define N_   2048
#define D_   64
#define QKV_ 192   // 3*D, rows: [0,64)=q, [64,128)=k, [128,192)=v

#define CHUNK_B   (QKV_ * 32 * 2)   // 12288 bytes per K-chunk weight slice
#define NBUF      4                 // quad buffer, prefetch distance 2
#define NCHUNK    (N_ / 32)         // 64 K-chunks

// ---------------------------------------------------------------------------
// CDNA5 async global->LDS copy (ASYNCcnt-tracked), per CDNA5 ISA.
// lds_off = byte offset in workgroup LDS (as3 offset), gaddr = 64-bit global.
// ---------------------------------------------------------------------------
__device__ __forceinline__ void async_b128(unsigned lds_off, const void* gaddr) {
  asm volatile("global_load_async_to_lds_b128 %0, %1, off"
               :: "v"(lds_off), "v"(gaddr) : "memory");
}
__device__ __forceinline__ void wait_async0() {
  asm volatile("s_wait_asynccnt 0x0" ::: "memory");
}
__device__ __forceinline__ void wait_async_le3() {
  asm volatile("s_wait_asynccnt 0x3" ::: "memory");
}

// ---------------------------------------------------------------------------
// Kernel 0: pack wq/wk/wv ([64][2048] f32 each) into bf16 Wt [192][2048]
// ---------------------------------------------------------------------------
__global__ __launch_bounds__(256)
void cvt_weights_kernel(const float* __restrict__ wq,
                        const float* __restrict__ wk,
                        const float* __restrict__ wv,
                        __bf16* __restrict__ Wt) {
  int t = blockIdx.x * 256 + threadIdx.x;
  if (t >= QKV_ * N_) return;
  int r = t / N_, k = t - r * N_;
  float v;
  if (r < 64)       v = wq[r * N_ + k];
  else if (r < 128) v = wk[(r - 64) * N_ + k];
  else              v = wv[(r - 128) * N_ + k];
  Wt[t] = (__bf16)v;
}

// ---------------------------------------------------------------------------
// Kernel 1: fused QKV GEMM.  qkv[b][m][0:192] = sum_k x[b][m][k] * Wt[c][k]
// grid = (N/128, B), block = 256 (8 waves). Wave w owns rows m0..m0+15,
// all 192 output columns (12 WMMA accumulator tiles of 16x16 f32).
// Weight slices (192x32 bf16 = 12 KB) are quad-buffered in LDS and
// prefetched 2 chunks ahead with global_load_async_to_lds_b128.  The loop
// body is branch-free: tail iterations re-fetch the last chunk into a
// dead buffer so the async issue and the asynccnt wait are unconditional.
// ---------------------------------------------------------------------------
__global__ __launch_bounds__(256)
void qkv_gemm_kernel(const float* __restrict__ x,     // [B][N][N] f32
                     const __bf16* __restrict__ Wt,   // [192][2048] bf16
                     float* __restrict__ qkv) {       // [B][N][192] f32
  __shared__ __bf16 Bs[NBUF][QKV_ * 32];              // 4 x 12 KB

  const int t  = threadIdx.x;
  const int b  = blockIdx.y;
  const int w  = t >> 5;
  const int l  = t & 31;
  const int m0 = blockIdx.x * 128 + w * 16;
  const int lh = (l < 16) ? 0 : 1;    // half-wave id
  const int lr = l & 15;

  // Async staging: each thread moves 3 x 16B of the 12 KB chunk.
  // Flat chunk offset o -> weight row r = o/64, byte col = o%64;
  // global addr advances 64B per K-chunk.
  const char* wt_g[3];
  unsigned    lds_o[3];
#pragma unroll
  for (int i = 0; i < 3; ++i) {
    int o = i * 4096 + t * 16;
    int r = o >> 6, c = o & 63;
    wt_g[i]  = (const char*)Wt + (size_t)r * (N_ * 2) + c;
    lds_o[i] = (unsigned)o;
  }
  const unsigned bs_base = (unsigned)(uintptr_t)&Bs[0][0];  // as3 byte offset

  v8f acc[12];
#pragma unroll
  for (int c = 0; c < 12; ++c) acc[c] = (v8f){0, 0, 0, 0, 0, 0, 0, 0};

  // A-source: row m0+lr of x[b]; contiguous K layout matching B: lanes 0-15
  // hold K {0..15}, lanes 16-31 hold K {16..31}.
  const float* aptr = x + ((size_t)b * N_ + (m0 + lr)) * (size_t)N_ + lh * 16;

  // Prologue: stage chunks 0 and 1 into buffers 0 and 1.
#pragma unroll
  for (int i = 0; i < 3; ++i) async_b128(bs_base + lds_o[i], wt_g[i]);
#pragma unroll
  for (int i = 0; i < 3; ++i)
    async_b128(bs_base + CHUNK_B + lds_o[i], wt_g[i] + 64);
  wait_async_le3();    // chunk 0 landed (chunk 1 may still be in flight)
  __syncthreads();

#pragma unroll 4
  for (int cc = 0; cc < NCHUNK; ++cc) {
    const int rb = cc & 3;              // read buffer (constant per body)
    const int wb = (cc + 2) & 3;        // write buffer (constant per body)

    // Unconditional prefetch of chunk min(cc+2, 63): tail re-fetches the
    // last chunk into a buffer that is never read again (harmless).
    int pc = cc + 2; if (pc > NCHUNK - 1) pc = NCHUNK - 1;
    {
      const unsigned dst = bs_base + (unsigned)wb * CHUNK_B;
#pragma unroll
      for (int i = 0; i < 3; ++i)
        async_b128(dst + lds_o[i], wt_g[i] + (size_t)pc * 64);
    }

    // A fragment: streamed x, non-temporal (read-once, keep L2 for weights/qkv)
    const float* ap = aptr + cc * 32;
    f32x4 a0 = __builtin_nontemporal_load((const f32x4*)(ap));
    f32x4 a1 = __builtin_nontemporal_load((const f32x4*)(ap + 4));
    f32x4 a2 = __builtin_nontemporal_load((const f32x4*)(ap + 8));
    f32x4 a3 = __builtin_nontemporal_load((const f32x4*)(ap + 12));
    v16bf A;
#pragma unroll
    for (int i = 0; i < 4; ++i) {
      A[i]      = (__bf16)a0[i];
      A[4 + i]  = (__bf16)a1[i];
      A[8 + i]  = (__bf16)a2[i];
      A[12 + i] = (__bf16)a3[i];
    }

    // 12 column tiles in 2 groups of 6: batch the ds_loads into distinct
    // registers so dscnt waits cover whole groups and loads co-execute
    // under earlier WMMAs.
    const __bf16* bcur = &Bs[0][0] + rb * (QKV_ * 32);
#pragma unroll
    for (int g = 0; g < 2; ++g) {
      v16bf Bf[6];
#pragma unroll
      for (int c2 = 0; c2 < 6; ++c2)
        Bf[c2] = *(const v16bf*)(bcur + ((g * 6 + c2) * 16 + lr) * 32 + lh * 16);
#pragma unroll
      for (int c2 = 0; c2 < 6; ++c2)
        acc[g * 6 + c2] = __builtin_amdgcn_wmma_f32_16x16x32_bf16(
            false, A, false, Bf[c2], (short)0, acc[g * 6 + c2], false, false);
    }

    // Only chunk cc+1 must have landed before the next iteration reads it;
    // the 3 newest async loads (chunk cc+2) may remain in flight.
    wait_async_le3();
    __syncthreads();
  }
  wait_async0();   // drain the dead tail prefetches

  // Store: C/D layout: VGPR j -> row j (lanes 0-15) / row j+8 (lanes 16-31),
  // column = lr within the 16-wide tile.  qkv stays L2-resident for kernel 2.
  float* out = qkv + ((size_t)b * N_ + m0) * QKV_;
#pragma unroll
  for (int c = 0; c < 12; ++c) {
#pragma unroll
    for (int j = 0; j < 8; ++j) {
      out[(size_t)(lh * 8 + j) * QKV_ + c * 16 + lr] = acc[c][j];
    }
  }
}

// ---------------------------------------------------------------------------
// Kernel 2: gather + dot + scale + broadcast.  One wave per (b,n) row.
// out[b,n,d] = (dot(q[b,n,:], k[b,key_ids[b,n],:]) / 8) * v[b,n,d]
// ---------------------------------------------------------------------------
__global__ __launch_bounds__(256)
void epilogue_kernel(const float* __restrict__ qkv,     // [B][N][192]
                     const int* __restrict__ key_ids,   // [B][N]
                     float* __restrict__ out) {         // [B][N][64]
  const int w = threadIdx.x >> 5;
  const int l = threadIdx.x & 31;
  const int r = blockIdx.x * 8 + w;        // flat (b,n)
  const int b = r >> 11;
  const int n = r & (N_ - 1);

  const int kid = key_ids[b * N_ + n];
  const float* qp = qkv + ((size_t)b * N_ + n)   * QKV_;        // q row
  const float* kp = qkv + ((size_t)b * N_ + kid) * QKV_ + 64;   // gathered k
  const float* vp = qp + 128;                                    // v row

  f32x2 q2 = *(const f32x2*)(qp + 2 * l);
  f32x2 k2 = *(const f32x2*)(kp + 2 * l);
  float p = q2[0] * k2[0] + q2[1] * k2[1];
#pragma unroll
  for (int off = 16; off > 0; off >>= 1) p += __shfl_xor(p, off, 32);
  float s = p * 0.125f;   // 1/sqrt(64)

  f32x2 v2 = *(const f32x2*)(vp + 2 * l);
  f32x2 o;
  o[0] = s * v2[0];
  o[1] = s * v2[1];
  *(f32x2*)(out + ((size_t)b * N_ + n) * (size_t)D_ + 2 * l) = o;
}

// ---------------------------------------------------------------------------
extern "C" void kernel_launch(void* const* d_in, const int* in_sizes, int n_in,
                              void* d_out, int out_size, void* d_ws, size_t ws_size,
                              hipStream_t stream) {
  const float* x       = (const float*)d_in[0];
  const float* wq      = (const float*)d_in[1];
  const float* wk      = (const float*)d_in[2];
  const float* wv      = (const float*)d_in[3];
  const int*   key_ids = (const int*)d_in[4];
  float*       out     = (float*)d_out;

  // workspace layout: [Wt bf16 192*2048][qkv f32 B*N*192]
  size_t wt_bytes = (size_t)QKV_ * N_ * sizeof(unsigned short);
  wt_bytes = (wt_bytes + 255) & ~(size_t)255;
  __bf16* Wt  = (__bf16*)d_ws;
  float*  qkv = (float*)((char*)d_ws + wt_bytes);

  cvt_weights_kernel<<<(QKV_ * N_ + 255) / 256, 256, 0, stream>>>(wq, wk, wv, Wt);
  qkv_gemm_kernel<<<dim3(N_ / 128, B_), 256, 0, stream>>>(x, Wt, qkv);
  epilogue_kernel<<<(B_ * N_) / 8, 256, 0, stream>>>(qkv, key_ids, out);
}
